// InvariantDescriptorBuilder_28011776704736
// MI455X (gfx1250) — matrix operand order repaired and checked by reading, not compiled, not run on hardware
//
#include <hip/hip_runtime.h>

// ---------------------------------------------------------------------------
// InvariantDescriptorBuilder for MI455X (gfx1250, wave32, WMMA).
//
// One wave32 == one atom.  Triplet channels A_{l,w} = F^T * M_{l,w} * F are
// computed with V_WMMA_F32_16X16X32_F16 (f16 operands, f32 accumulate):
//   stage 1:  T = Ft(16x32, rows 8..15 zero) x M(32x32)   -> 2 WMMAs
//   stage 2:  A = T(16x32) x F(32x16, cols 8..15 zero)    -> 1 WMMA
// 9 channels (l in 0..2, w in {1, m_j.m_k, s_j*s_k}) -> 27 WMMAs / atom.
//
// Pair-invariant data (cos, masked weights) is hoisted into packed-f16
// registers in B-operand element order, so the per-channel M build is just
// v_pk_mul_f16 / v_pk_fma_f16 — no repeated LDS reads or dot products.
// Scalar channels (rho_r, rho_uj, rho_s) use fp32 wave32 shuffle reductions.
// ---------------------------------------------------------------------------

typedef __attribute__((ext_vector_type(16))) _Float16 v16h;
typedef __attribute__((ext_vector_type(8)))  float    v8f;

union V16H { v16h v; _Float16 e[16]; };
union V8F  { v8f  v; float    e[8];  };

#define K_NEIGH 32
#define R_RAD   8
#define WAVES_PER_BLOCK 8
#define NDESC 603

// 16-bit A-operand layout (16x32 MxK), contiguous K to match B:
//   M=lane%16, lanes 0-15 cover K=0..15 (K=2r+s), lanes 16-31 K=16..31.
__device__ __forceinline__ int kidxA(int r, int s, int hi) {
    return 16 * hi + 2 * r + s;
}
// 16-bit B-operand k order used here (32x16 KxN), contiguous K to match A:
//   lane = column N, lanes 0-15 cover K=0..15 (K=2r+s), lanes 16-31 K=16..31.
__device__ __forceinline__ int kidxB(int r, int s, int hi) {
    return 16 * hi + 2 * r + s;
}

__global__ __launch_bounds__(256)
void invariant_descriptor_kernel(const float* __restrict__ pos,
                                 const float* __restrict__ mag,
                                 const int*   __restrict__ nbr,
                                 float* __restrict__ out,
                                 int nAtoms)
{
    constexpr float CUTOFF   = 4.5f;
    constexpr float SPACING  = CUTOFF / (R_RAD - 1);
    constexpr float BETA     = 1.0f / (SPACING * SPACING);
    constexpr float U_REF    = 2.2f * 2.2f;
    constexpr float INV_UREF = 1.0f / U_REF;
    constexpr float INV_SQRT_UREF = 1.0f / 2.2f;
    constexpr float EPS      = 1e-8f;
    constexpr float PI_F     = 3.14159265358979f;

    __shared__ float    sQ[WAVES_PER_BLOCK][K_NEIGH][3];      // unit vectors
    __shared__ float    sM[WAVES_PER_BLOCK][K_NEIGH][3];      // mag_j / sqrt(U_REF)
    __shared__ float    sS[WAVES_PER_BLOCK][K_NEIGH];         // (m_i.m_j)/U_REF
    __shared__ _Float16 sF[WAVES_PER_BLOCK][K_NEIGH][R_RAD];  // radial basis (f16)
    __shared__ float    sT[WAVES_PER_BLOCK][16][K_NEIGH];     // stage-1 relayout

    const int wv   = threadIdx.x >> 5;
    const int lane = threadIdx.x & 31;
    const int atom = blockIdx.x * WAVES_PER_BLOCK + wv;
    if (atom >= nAtoms) return;           // wave-uniform -> EXEC stays all-ones

    const int m16 = lane & 15;
    const int hi  = lane >> 4;

    // ------------------- gather phase: lane == neighbor j -------------------
    const int j  = lane;
    const int nj = nbr[atom * K_NEIGH + j];
    // prefetch the next block-row of the neighbor table (global_prefetch_b8)
    __builtin_prefetch(&nbr[(atom + WAVES_PER_BLOCK) * K_NEIGH + j], 0, 3);

    const float pix = pos[3 * atom + 0], piy = pos[3 * atom + 1], piz = pos[3 * atom + 2];
    const float mix = mag[3 * atom + 0], miy = mag[3 * atom + 1], miz = mag[3 * atom + 2];

    const float vx = pos[3 * nj + 0] - pix;
    const float vy = pos[3 * nj + 1] - piy;
    const float vz = pos[3 * nj + 2] - piz;
    const float d     = sqrtf(vx * vx + vy * vy + vz * vz + EPS);
    const float inv_d = 1.0f / d;

    const float mjx = mag[3 * nj + 0], mjy = mag[3 * nj + 1], mjz = mag[3 * nj + 2];
    const float uj = (mjx * mjx + mjy * mjy + mjz * mjz) * INV_UREF;
    const float sj = (mix * mjx + miy * mjy + miz * mjz) * INV_UREF;

    // radial basis with cosine cutoff
    float xc = fminf(PI_F * d * (1.0f / CUTOFF), PI_F);
    float fcut = 0.5f * (__cosf(xc) + 1.0f);
    fcut = (d < CUTOFF) ? fcut : 0.0f;

    float f[R_RAD];
    #pragma unroll
    for (int a = 0; a < R_RAD; ++a) {
        const float diff = d - (float)a * SPACING;
        f[a] = __expf(-BETA * diff * diff) * fcut;
    }

    sQ[wv][j][0] = vx * inv_d;
    sQ[wv][j][1] = vy * inv_d;
    sQ[wv][j][2] = vz * inv_d;
    sM[wv][j][0] = mjx * INV_SQRT_UREF;
    sM[wv][j][1] = mjy * INV_SQRT_UREF;
    sM[wv][j][2] = mjz * INV_SQRT_UREF;
    sS[wv][j]    = sj;
    #pragma unroll
    for (int a = 0; a < R_RAD; ++a) sF[wv][j][a] = (_Float16)f[a];

    // ------------- fp32 wave reductions: rho_r, rho_uj, rho_s --------------
    float* op = out + (size_t)atom * NDESC;
    #pragma unroll
    for (int a = 0; a < R_RAD; ++a) {
        float v0 = f[a], v1 = uj * f[a], v2 = sj * f[a];
        #pragma unroll
        for (int off = 16; off > 0; off >>= 1) {
            v0 += __shfl_xor(v0, off, 32);
            v1 += __shfl_xor(v1, off, 32);
            v2 += __shfl_xor(v2, off, 32);
        }
        if (lane == 0) {
            op[3 + a]   = v0;   // rho_r
            op[203 + a] = v1;   // rho_uj
            op[211 + a] = v2;   // rho_s
        }
    }
    if (lane == 0) {
        const float un = (mix * mix + miy * miy + miz * miz) * INV_UREF;
        op[0] = un;
        op[1] = un * un;
        op[2] = un * un * un;
    }

    // --------- channel-invariant WMMA operands built from LDS f16 ----------
    // af: stage-1 A operand = F^T padded to 16x32 (rows 8..15 zero)
    // bf: stage-2 B operand = F   padded to 32x16 (cols 8..15 zero)
    V16H af, bf;
    #pragma unroll
    for (int r = 0; r < 8; ++r) {
        #pragma unroll
        for (int s = 0; s < 2; ++s) {
            const int ja = kidxA(r, s, hi);
            const int jb = kidxB(r, s, hi);
            af.e[2 * r + s] = (m16 < 8) ? sF[wv][ja][m16] : (_Float16)0.0f;
            bf.e[2 * r + s] = (m16 < 8) ? sF[wv][jb][m16] : (_Float16)0.0f;
        }
    }

    // --------- hoist pair-invariant data into packed f16 registers ---------
    // Element order matches the B operand: half h, element p = 2r+s maps to
    // pair (j = kidxB(r,s,hi), k = m16 + 16h).  Masks (j<k) folded into w*.
    V16H cp[2];   // clamped cos(theta_jk)
    V16H w0p[2];  // mask
    V16H w1p[2];  // mask * (m_j . m_k)/U_REF
    V16H w2p[2];  // mask * s_j * s_k

    float qn[2][3], mn[2][3], snv[2];
    #pragma unroll
    for (int h = 0; h < 2; ++h) {
        const int n = m16 + 16 * h;
        qn[h][0] = sQ[wv][n][0]; qn[h][1] = sQ[wv][n][1]; qn[h][2] = sQ[wv][n][2];
        mn[h][0] = sM[wv][n][0]; mn[h][1] = sM[wv][n][1]; mn[h][2] = sM[wv][n][2];
        snv[h]   = sS[wv][n];
    }
    #pragma unroll
    for (int r = 0; r < 8; ++r) {
        #pragma unroll
        for (int s = 0; s < 2; ++s) {
            const int jj = kidxB(r, s, hi);
            const float q0 = sQ[wv][jj][0], q1 = sQ[wv][jj][1], q2 = sQ[wv][jj][2];
            const float a0 = sM[wv][jj][0], a1 = sM[wv][jj][1], a2v = sM[wv][jj][2];
            const float sv = sS[wv][jj];
            #pragma unroll
            for (int h = 0; h < 2; ++h) {
                const int p = 2 * r + s;
                const bool msk = jj < (m16 + 16 * h);
                float c = q0 * qn[h][0] + q1 * qn[h][1] + q2 * qn[h][2];
                c = fminf(fmaxf(c, -1.0f), 1.0f);
                cp[h].e[p]  = (_Float16)c;
                w0p[h].e[p] = msk ? (_Float16)1.0f : (_Float16)0.0f;
                w1p[h].e[p] = msk ? (_Float16)(a0 * mn[h][0] + a1 * mn[h][1] + a2v * mn[h][2])
                                  : (_Float16)0.0f;
                w2p[h].e[p] = msk ? (_Float16)(sv * snv[h]) : (_Float16)0.0f;
            }
        }
    }

    const v8f zero = {0.0f, 0.0f, 0.0f, 0.0f, 0.0f, 0.0f, 0.0f, 0.0f};

    // ------------------- 9 triplet channels, 3 WMMAs each ------------------
    #pragma unroll
    for (int l = 0; l <= 2; ++l) {
        // packed Legendre values per half (f16), shared by the 3 weights
        V16H Pp[2];
        #pragma unroll
        for (int h = 0; h < 2; ++h) {
            #pragma unroll
            for (int p = 0; p < 16; ++p) {
                const _Float16 c = cp[h].e[p];
                Pp[h].e[p] = (l == 1) ? c
                           : (_Float16)((_Float16)1.5f * c * c - (_Float16)0.5f);
            }
        }

        #pragma unroll
        for (int wsel = 0; wsel < 3; ++wsel) {
            const V16H* wp = (wsel == 0) ? w0p : (wsel == 1) ? w1p : w2p;

            // B operand halves: elementwise packed-f16 multiply (masked in w)
            V16H b0, b1;
            #pragma unroll
            for (int p = 0; p < 16; ++p) {
                b0.e[p] = (l == 0) ? wp[0].e[p] : Pp[0].e[p] * wp[0].e[p];
                b1.e[p] = (l == 0) ? wp[1].e[p] : Pp[1].e[p] * wp[1].e[p];
            }

            // Stage 1: T = Ft x M   (two 16x16x32 WMMAs, f32 accumulate)
            V8F d0, d1;
            d0.v = __builtin_amdgcn_wmma_f32_16x16x32_f16(
                       false, af.v, false, b0.v, (short)0, zero, false, false);
            d1.v = __builtin_amdgcn_wmma_f32_16x16x32_f16(
                       false, af.v, false, b1.v, (short)0, zero, false, false);

            // Relayout C/D tile -> A-operand layout through LDS (intra-wave).
            #pragma unroll
            for (int v = 0; v < 8; ++v) {
                sT[wv][v + 8 * hi][m16]      = d0.e[v];
                sT[wv][v + 8 * hi][m16 + 16] = d1.e[v];
            }

            V16H a2;
            #pragma unroll
            for (int r = 0; r < 8; ++r) {
                #pragma unroll
                for (int s = 0; s < 2; ++s)
                    a2.e[2 * r + s] = (_Float16)sT[wv][m16][kidxA(r, s, hi)];
            }

            // Stage 2: A = T x F   (one 16x16x32 WMMA)
            V8F d2;
            d2.v = __builtin_amdgcn_wmma_f32_16x16x32_f16(
                       false, a2.v, false, bf.v, (short)0, zero, false, false);

            // Write the valid 8x8 block: row a = VGPR index, col b = lane.
            // Descriptor layout: index = base + (a*8 + b)*3 + l.
            const int base = (wsel == 0) ? 11 : (wsel == 1) ? 219 : 411;
            if (hi == 0 && m16 < 8) {
                #pragma unroll
                for (int v = 0; v < 8; ++v)
                    op[base + (v * 8 + m16) * 3 + l] = d2.e[v];
            }
        }
    }
}

extern "C" void kernel_launch(void* const* d_in, const int* in_sizes, int n_in,
                              void* d_out, int out_size, void* d_ws, size_t ws_size,
                              hipStream_t stream) {
    const float* pos  = (const float*)d_in[0];
    const float* mag  = (const float*)d_in[1];
    const int*   nidx = (const int*)d_in[2];
    float*       out  = (float*)d_out;

    const int nAtoms = in_sizes[2] / K_NEIGH;   // 20000
    const int blocks = (nAtoms + WAVES_PER_BLOCK - 1) / WAVES_PER_BLOCK;

    invariant_descriptor_kernel<<<blocks, WAVES_PER_BLOCK * 32, 0, stream>>>(
        pos, mag, nidx, out, nAtoms);
}
